// DividedSpaceTimeAttention_26053271618173
// MI455X (gfx1250) — matrix operand order repaired and checked by reading, not compiled, not run on hardware
//
#include <hip/hip_runtime.h>
#include <hip/hip_bf16.h>

// ---------------------------------------------------------------------------
// Types for CDNA5 WMMA (wave32, V_WMMA_F32_16X16X32_BF16)
// ---------------------------------------------------------------------------
typedef __attribute__((ext_vector_type(16))) __bf16 v16bf;
typedef __attribute__((ext_vector_type(8)))  float  v8f;
typedef __attribute__((ext_vector_type(4)))  unsigned uint4v;

struct Q2     { uint4v a, b; };            // 32 bytes -> v16bf
struct U16x16 { unsigned short u[16]; };   // 32 bytes -> v16bf
struct U16x8  { unsigned short u[8]; };    // 16 bytes -> uint4v

__device__ __forceinline__ unsigned short f2bf(float f) {
    unsigned u = __float_as_uint(f);
    u += 0x7fffu + ((u >> 16) & 1u);       // round-to-nearest-even
    return (unsigned short)(u >> 16);
}

// A fragment: A is (M=16, K=32) row-major bf16, `A` pre-offset to (row0, k0).
// ISA layout: lanes 0-15 = rows, hold K {0..7,16..23}; lanes 16-31 hold K {8..15,24..31}.
__device__ __forceinline__ v16bf load_a_frag(const unsigned short* A, size_t lda, int lane) {
    int m    = lane & 15;
    int koff = (lane >> 4) << 3;           // 0 or 8
    const unsigned short* p = A + (size_t)m * lda + koff;
    Q2 q;
    q.a = *(const uint4v*)(p);             // K koff..koff+7
    q.b = *(const uint4v*)(p + 16);        // K koff+16..koff+23
    return __builtin_bit_cast(v16bf, q);
}

// B fragment: Bt is (N=16, K=32) row-major bf16 (i.e. B transposed), pre-offset (col0, k0).
// ISA layout: lanes 0-15 = col n with K {0..7,16..23}; lanes 16-31 = col n with K {8..15,24..31}.
__device__ __forceinline__ v16bf load_b_frag(const unsigned short* Bt, size_t ldb, int lane) {
    int n  = lane & 15;
    int kb = (lane >> 4) << 3;             // 0 or 8
    const unsigned short* p = Bt + (size_t)n * ldb + kb;
    Q2 q;
    q.a = *(const uint4v*)(p);
    q.b = *(const uint4v*)(p + 16);
    return __builtin_bit_cast(v16bf, q);
}

#define WMMA_BF16(a, b, c) \
    __builtin_amdgcn_wmma_f32_16x16x32_bf16(false, (a), false, (b), (short)0, (c), false, false)

// ---------------------------------------------------------------------------
// Problem constants
// ---------------------------------------------------------------------------
// B=2 C=256 T=16 H=32 W=32  -> HW=1024, frames = B*T = 32, temporal rows R = B*H*W*T = 32768
#define HWSZ   1024
#define CCH    256
#define NFRM   32
#define FRAME_ELEMS 262144ull      /* C*HW per frame */
#define RROWS  32768

// ---------------------------------------------------------------------------
// Generic batched bf16 WMMA GEMM: D = A(M,K) * Bt(N,K)^T, fp32 accumulate.
// Each wave computes a 16(M) x 64(N) tile (4 accumulators). blockDim = 256 (8 waves).
// modes: 0 bf16 natural (+bias[m])       1 bf16 transposed store (+bias[m])
//        2 f32 natural                   3 f32: + bias[m] + resid, into d_out (spatial proj)
//        4 f32 scatter-add temporal:  d_out[b, n, t, hw] += acc + bias[n]
// ---------------------------------------------------------------------------
__global__ __launch_bounds__(256)
void gemm_bf16_wmma(const unsigned short* __restrict__ A,
                    const unsigned short* __restrict__ Bt,
                    void* __restrict__ D,
                    const float* __restrict__ bias,
                    const float* __restrict__ resid,
                    int M, int N, int K, int lda, int ldb, int ldd,
                    size_t sA, size_t sB, size_t sD, int mode)
{
    size_t zb = blockIdx.z;
    A  += zb * sA;
    Bt += zb * sB;
    size_t doff = zb * sD;

    int lane = threadIdx.x & 31;
    int wv   = threadIdx.x >> 5;
    int tiles_n = N >> 6;
    int tile_id = blockIdx.x * 8 + wv;
    if (tile_id >= (M >> 4) * tiles_n) return;
    int tm = tile_id / tiles_n;
    int tn = tile_id - tm * tiles_n;
    int row0 = tm << 4;
    int col0 = tn << 6;

    v8f acc[4] = {v8f{}, v8f{}, v8f{}, v8f{}};
    const unsigned short* Ab = A  + (size_t)row0 * lda;
    const unsigned short* Bb = Bt + (size_t)col0 * ldb;

    for (int k0 = 0; k0 < K; k0 += 32) {
        v16bf a  = load_a_frag(Ab + k0, lda, lane);
        v16bf b0 = load_b_frag(Bb + k0,                       ldb, lane);
        v16bf b1 = load_b_frag(Bb + (size_t)16 * ldb + k0,    ldb, lane);
        v16bf b2 = load_b_frag(Bb + (size_t)32 * ldb + k0,    ldb, lane);
        v16bf b3 = load_b_frag(Bb + (size_t)48 * ldb + k0,    ldb, lane);
        acc[0] = WMMA_BF16(a, b0, acc[0]);
        acc[1] = WMMA_BF16(a, b1, acc[1]);
        acc[2] = WMMA_BF16(a, b2, acc[2]);
        acc[3] = WMMA_BF16(a, b3, acc[3]);
    }

    int nloc = lane & 15;
    int hs   = (lane >> 4) << 3;           // D layout: lanes>=16 hold rows m+8

    #pragma unroll
    for (int t = 0; t < 4; ++t) {
        int col = col0 + t * 16 + nloc;
        if (mode == 0) {                                   // bf16 natural (M,N)
            unsigned short* dst = (unsigned short*)D + doff;
            #pragma unroll
            for (int r = 0; r < 8; ++r) {
                int m = row0 + r + hs;
                float v = acc[t][r] + (bias ? bias[m] : 0.f);
                dst[(size_t)m * ldd + col] = f2bf(v);
            }
        } else if (mode == 1) {                            // bf16 transposed (N,M)
            unsigned short* dst = (unsigned short*)D + doff;
            U16x8 pk;
            #pragma unroll
            for (int r = 0; r < 8; ++r) {
                int m = row0 + r + hs;
                pk.u[r] = f2bf(acc[t][r] + (bias ? bias[m] : 0.f));
            }
            *(uint4v*)(dst + (size_t)col * ldd + row0 + hs) = __builtin_bit_cast(uint4v, pk);
        } else if (mode == 2) {                            // f32 natural
            float* dst = (float*)D + doff;
            #pragma unroll
            for (int r = 0; r < 8; ++r)
                dst[(size_t)(row0 + r + hs) * ldd + col] = acc[t][r];
        } else if (mode == 3) {                            // spatial proj epilogue
            float* dst = (float*)D + doff;
            const float* rs = resid + doff;
            #pragma unroll
            for (int r = 0; r < 8; ++r) {
                int m = row0 + r + hs;
                size_t idx = (size_t)m * ldd + col;
                dst[idx] = acc[t][r] + bias[m] + rs[idx];
            }
        } else {                                           // temporal scatter-add
            float* dst = (float*)D;
            #pragma unroll
            for (int r = 0; r < 8; ++r) {
                int m = row0 + r + hs;          // row = ((b*HW + hw)*16 + t)
                int tt = m & 15;
                size_t bhw = (size_t)m >> 4;
                size_t b   = bhw >> 10;
                size_t hw  = bhw & 1023;
                size_t idx = b * 4194304ull + (size_t)col * 16384ull
                           + (size_t)tt * 1024ull + hw;
                dst[idx] += acc[t][r] + bias[col];
            }
        }
    }
}

// ---------------------------------------------------------------------------
// GroupNorm over the raw-reshaped (32, 256, 1024) spatial view of x.
// One block per (frame, group of 8 channels); writes hfT (frame, hw, c) bf16.
// ---------------------------------------------------------------------------
__global__ __launch_bounds__(256)
void groupnorm_kernel(const float* __restrict__ x,
                      const float* __restrict__ gamma,
                      const float* __restrict__ beta,
                      unsigned short* __restrict__ hfT)
{
    int f = blockIdx.x >> 5;
    int g = blockIdx.x & 31;
    size_t xb = (size_t)f * FRAME_ELEMS + (size_t)g * 8192;
    int tid = threadIdx.x;

    float s0 = 0.f, s1 = 0.f;
    for (int i = tid; i < 8192; i += 256) {
        float v = x[xb + i];
        s0 += v; s1 += v * v;
    }
    __shared__ float sh0[256], sh1[256];
    sh0[tid] = s0; sh1[tid] = s1; __syncthreads();
    for (int st = 128; st > 0; st >>= 1) {
        if (tid < st) { sh0[tid] += sh0[tid + st]; sh1[tid] += sh1[tid + st]; }
        __syncthreads();
    }
    float mean = sh0[0] * (1.f / 8192.f);
    float var  = sh1[0] * (1.f / 8192.f) - mean * mean;
    float rstd = rsqrtf(var + 1e-6f);

    for (int i = tid; i < 8192; i += 256) {
        int cl = i >> 10, hw = i & 1023;
        int c  = (g << 3) + cl;
        float v = (x[xb + i] - mean) * rstd * gamma[c] + beta[c];
        hfT[(size_t)f * FRAME_ELEMS + (size_t)hw * CCH + c] = f2bf(v);
    }
}

// Row softmax (row length 1024): P = softmax(S * scale), bf16 out.
__global__ __launch_bounds__(256)
void softmax_kernel(const float* __restrict__ S, unsigned short* __restrict__ P, float scale)
{
    size_t row = blockIdx.x;
    const float* src = S + row * 1024;
    unsigned short* dst = P + row * 1024;
    int tid = threadIdx.x;

    float v[4]; float mx = -3.0e38f;
    #pragma unroll
    for (int j = 0; j < 4; ++j) {
        v[j] = src[tid + j * 256] * scale;
        mx = fmaxf(mx, v[j]);
    }
    __shared__ float sh[256];
    sh[tid] = mx; __syncthreads();
    for (int st = 128; st > 0; st >>= 1) {
        if (tid < st) sh[tid] = fmaxf(sh[tid], sh[tid + st]);
        __syncthreads();
    }
    float m = sh[0]; __syncthreads();

    float sm = 0.f;
    #pragma unroll
    for (int j = 0; j < 4; ++j) { v[j] = __expf(v[j] - m); sm += v[j]; }
    sh[tid] = sm; __syncthreads();
    for (int st = 128; st > 0; st >>= 1) {
        if (tid < st) sh[tid] += sh[tid + st];
        __syncthreads();
    }
    float inv = 1.f / sh[0];
    #pragma unroll
    for (int j = 0; j < 4; ++j) dst[tid + j * 256] = f2bf(v[j] * inv);
}

// Weight convert to bf16, optionally transposed. n = 256*256, grid 256x256.
__global__ __launch_bounds__(256)
void convert_w(const float* __restrict__ src, unsigned short* __restrict__ dst, int transpose)
{
    int i = blockIdx.x * 256 + threadIdx.x;
    int r = i >> 8, c = i & 255;
    float v = transpose ? src[c * 256 + r] : src[i];
    dst[i] = f2bf(v);
}

// x (B,C,T,H,W) f32 -> xt ((b,hw,t) rows, C cols) bf16
__global__ __launch_bounds__(256)
void xt_convert(const float* __restrict__ x, unsigned short* __restrict__ xt)
{
    size_t i = (size_t)blockIdx.x * 256 + threadIdx.x;   // 8388608 elements
    size_t hw = i & 1023;
    size_t t  = (i >> 10) & 15;
    size_t c  = (i >> 14) & 255;
    size_t b  = i >> 22;
    size_t r  = ((b << 10) | hw) * 16 + t;
    xt[r * 256 + c] = f2bf(x[i]);
}

// ---------------------------------------------------------------------------
// Temporal axial attention core: one wave per (b,h,w,head).
// QK^T via WMMA, cross-lane softmax (rows live in one VGPR across a 16-lane
// half in the D layout), P reshuffled through LDS into A-fragment layout,
// then P * V via two WMMAs (K padded 16->32 with zeros).
// ---------------------------------------------------------------------------
__global__ __launch_bounds__(256)
void temporal_attn(const unsigned short* __restrict__ qb,
                   const unsigned short* __restrict__ kb,
                   const unsigned short* __restrict__ vb,
                   unsigned short* __restrict__ outh)
{
    __shared__ __align__(16) unsigned short pS[8][16 * 32];
    int lane = threadIdx.x & 31;
    int wv   = threadIdx.x >> 5;
    int gw   = blockIdx.x * 8 + wv;        // 0 .. 16383
    int head = gw & 7;
    size_t rb = (size_t)(gw >> 3) << 4;    // row base = (b*HW + hw) * T
    int cb = head << 5;                    // channel base = head * 32

    v16bf a = load_a_frag(qb + rb * 256 + cb, 256, lane);
    v16bf b = load_b_frag(kb + rb * 256 + cb, 256, lane);
    v8f s = {};
    s = WMMA_BF16(a, b, s);

    float p[8];
    #pragma unroll
    for (int r = 0; r < 8; ++r) {
        float xv = s[r] * 0.17677669529663687f;   // 1/sqrt(dk=32)
        float mx = xv;
        mx = fmaxf(mx, __shfl_xor(mx, 1));
        mx = fmaxf(mx, __shfl_xor(mx, 2));
        mx = fmaxf(mx, __shfl_xor(mx, 4));
        mx = fmaxf(mx, __shfl_xor(mx, 8));
        float e = __expf(xv - mx);
        float sm = e;
        sm += __shfl_xor(sm, 1);
        sm += __shfl_xor(sm, 2);
        sm += __shfl_xor(sm, 4);
        sm += __shfl_xor(sm, 8);
        p[r] = e / sm;
    }

    int nloc = lane & 15;
    int hs   = (lane >> 4) << 3;
    #pragma unroll
    for (int r = 0; r < 8; ++r) {
        int t = r + hs;
        pS[wv][t * 32 + nloc]      = f2bf(p[r]);
        pS[wv][t * 32 + 16 + nloc] = 0;            // zero pad K 16..31
    }
    __syncthreads();

    v16bf pa = load_a_frag(&pS[wv][0], 32, lane);

    U16x16 bv0 = {}, bv1 = {};
    {                                              // elements 8..15 cover K 16..31 -> all zero
        #pragma unroll
        for (int j = 0; j < 8; ++j) {
            size_t base = (rb + hs + j) * 256 + cb;
            bv0.u[j] = vb[base + nloc];
            bv1.u[j] = vb[base + 16 + nloc];
        }
    }
    v8f o0 = {}, o1 = {};
    o0 = WMMA_BF16(pa, __builtin_bit_cast(v16bf, bv0), o0);
    o1 = WMMA_BF16(pa, __builtin_bit_cast(v16bf, bv1), o1);

    #pragma unroll
    for (int r = 0; r < 8; ++r) {
        int t = r + hs;
        size_t base = (rb + t) * 256 + cb;
        outh[base + nloc]      = f2bf(o0[r]);
        outh[base + 16 + nloc] = f2bf(o1[r]);
    }
}

// ---------------------------------------------------------------------------
// Host orchestration
// ---------------------------------------------------------------------------
extern "C" void kernel_launch(void* const* d_in, const int* in_sizes, int n_in,
                              void* d_out, int out_size, void* d_ws, size_t ws_size,
                              hipStream_t stream)
{
    (void)in_sizes; (void)n_in; (void)out_size;
    const float* x    = (const float*)d_in[0];
    const float* gng  = (const float*)d_in[1];
    const float* gnb  = (const float*)d_in[2];
    const float* wq_s = (const float*)d_in[3];
    const float* bq_s = (const float*)d_in[4];
    const float* wk_s = (const float*)d_in[5];
    const float* bk_s = (const float*)d_in[6];
    const float* wv_s = (const float*)d_in[7];
    const float* bv_s = (const float*)d_in[8];
    const float* wo_s = (const float*)d_in[9];
    const float* bo_s = (const float*)d_in[10];
    const float* wq_t = (const float*)d_in[11];
    const float* wk_t = (const float*)d_in[12];
    const float* wv_t = (const float*)d_in[13];
    const float* wf_t = (const float*)d_in[14];
    const float* bf_t = (const float*)d_in[15];
    float* out = (float*)d_out;
    char*  ws  = (char*)d_ws;

    unsigned short* W = (unsigned short*)ws;        // 8 weight mats * 65536 bf16
    char* big = ws + (1 << 20);
    const size_t FB = 16777216ull;                  // 32 frames * 1024 * 256 * 2B

    unsigned short* hfT = (unsigned short*)(big);
    unsigned short* qT  = (unsigned short*)(big + 1 * FB);
    unsigned short* kT  = (unsigned short*)(big + 2 * FB);
    unsigned short* vS  = (unsigned short*)(big + 3 * FB);
    unsigned short* oT  = (unsigned short*)(big + 4 * FB);
    char* tail = big + 5 * FB;

    int cf = 8;                                     // frames per score chunk
    while (cf > 1 &&
           (size_t)(1 << 20) + 5 * FB + (size_t)cf * (4194304ull + 2097152ull) > ws_size)
        cf >>= 1;
    float*          Sc = (float*)tail;
    unsigned short* Pm = (unsigned short*)(tail + (size_t)cf * 4194304ull);

    // temporal stage aliases the spatial buffers (spatial fully drains first)
    unsigned short* xt = (unsigned short*)(big);
    unsigned short* qb = (unsigned short*)(big + 1 * FB);
    unsigned short* kb = (unsigned short*)(big + 2 * FB);
    unsigned short* vb = (unsigned short*)(big + 3 * FB);
    unsigned short* oh = (unsigned short*)(big + 4 * FB);

    auto gemm = [&](const unsigned short* A, const unsigned short* Bt, void* D,
                    const float* bias, const float* resid,
                    int M, int N, int K, int lda, int ldb, int ldd,
                    size_t sA, size_t sB, size_t sD, int mode, int nz) {
        int tiles  = (M >> 4) * (N >> 6);
        int blocks = (tiles + 7) >> 3;
        gemm_bf16_wmma<<<dim3(blocks, 1, nz), 256, 0, stream>>>(
            A, Bt, D, bias, resid, M, N, K, lda, ldb, ldd, sA, sB, sD, mode);
    };

    // ---- weights -> bf16 (spatial natural, temporal transposed) ----
    convert_w<<<256, 256, 0, stream>>>(wq_s, W + 0 * 65536, 0);
    convert_w<<<256, 256, 0, stream>>>(wk_s, W + 1 * 65536, 0);
    convert_w<<<256, 256, 0, stream>>>(wv_s, W + 2 * 65536, 0);
    convert_w<<<256, 256, 0, stream>>>(wo_s, W + 3 * 65536, 0);
    convert_w<<<256, 256, 0, stream>>>(wq_t, W + 4 * 65536, 1);
    convert_w<<<256, 256, 0, stream>>>(wk_t, W + 5 * 65536, 1);
    convert_w<<<256, 256, 0, stream>>>(wv_t, W + 6 * 65536, 1);
    convert_w<<<256, 256, 0, stream>>>(wf_t, W + 7 * 65536, 1);

    // ---- spatial branch ----
    groupnorm_kernel<<<1024, 256, 0, stream>>>(x, gng, gnb, hfT);

    // q/k (stored transposed (HW,C)), v (natural (C,HW)); batched over 32 frames
    gemm(W + 0 * 65536, hfT, qT, bq_s, nullptr, 256, 1024, 256, 256, 256, 256,
         0, FRAME_ELEMS, FRAME_ELEMS, 1, 32);
    gemm(W + 1 * 65536, hfT, kT, bk_s, nullptr, 256, 1024, 256, 256, 256, 256,
         0, FRAME_ELEMS, FRAME_ELEMS, 1, 32);
    gemm(W + 2 * 65536, hfT, vS, bv_s, nullptr, 256, 1024, 256, 256, 256, 1024,
         0, FRAME_ELEMS, FRAME_ELEMS, 0, 32);

    // scores -> softmax -> out = V * P^T (stored transposed (HW,C)), chunked
    for (int f0 = 0; f0 < NFRM; f0 += cf) {
        gemm(qT + (size_t)f0 * FRAME_ELEMS, kT + (size_t)f0 * FRAME_ELEMS, Sc,
             nullptr, nullptr, 1024, 1024, 256, 256, 256, 1024,
             FRAME_ELEMS, FRAME_ELEMS, 1048576ull, 2, cf);
        softmax_kernel<<<cf * 1024, 256, 0, stream>>>(Sc, Pm, 0.0625f); // C^-0.5
        gemm(vS + (size_t)f0 * FRAME_ELEMS, Pm, oT + (size_t)f0 * FRAME_ELEMS,
             nullptr, nullptr, 256, 1024, 1024, 1024, 1024, 256,
             FRAME_ELEMS, 1048576ull, FRAME_ELEMS, 1, cf);
    }

    // proj + bias + residual -> d_out = x + spatial_out
    gemm(W + 3 * 65536, oT, out, bo_s, x, 256, 1024, 256, 256, 256, 1024,
         0, FRAME_ELEMS, FRAME_ELEMS, 3, 32);

    // ---- temporal branch (buffers alias spatial ones; stream-ordered) ----
    xt_convert<<<32768, 256, 0, stream>>>(x, xt);

    gemm(xt, W + 4 * 65536, qb, nullptr, nullptr, RROWS, 256, 256, 256, 256, 256,
         0, 0, 0, 0, 1);
    gemm(xt, W + 5 * 65536, kb, nullptr, nullptr, RROWS, 256, 256, 256, 256, 256,
         0, 0, 0, 0, 1);
    gemm(xt, W + 6 * 65536, vb, nullptr, nullptr, RROWS, 256, 256, 256, 256, 256,
         0, 0, 0, 0, 1);

    temporal_attn<<<2048, 256, 0, stream>>>(qb, kb, vb, oh);

    // tt = outh @ wf + bf, scatter-added into d_out
    gemm(oh, W + 7 * 65536, out, bf_t, nullptr, RROWS, 256, 256, 256, 256, 0,
         0, 0, 0, 4, 1);
}
